// QwenAttention_15333033247491
// MI455X (gfx1250) — matrix operand order repaired and checked by reading, not compiled, not run on hardware
//
#include <hip/hip_runtime.h>
#include <hip/hip_bf16.h>

// ---------------------------------------------------------------------------
// Qwen GQA attention block for gfx1250 (MI455X), bf16 WMMA pipeline.
// B=2, S=2048, H=2048, NH=16, NKV=2, HD=128, KV=256, theta=1e6.
// WMMA operand fragments load as contiguous 128-bit LDS chunks; all LDS
// staging uses gfx1250 async global->LDS copies (ASYNCcnt), double-buffered
// so the DMA of tile t+1 overlaps the WMMA work on tile t.
// ---------------------------------------------------------------------------

#define B_  2
#define S_  2048
#define H_  2048
#define NH_ 16
#define NKV_ 2
#define HD_ 128
#define KV_ (NKV_ * HD_)
#define MROWS_ (B_ * S_)   // 4096

typedef __attribute__((ext_vector_type(16))) __bf16 v16bf;
typedef __attribute__((ext_vector_type(8)))  float  v8f;

// ------------------------------ helpers ------------------------------------

__device__ __forceinline__ unsigned short f32_to_bf16_bits(float f) {
  unsigned int x = __builtin_bit_cast(unsigned int, f);
  unsigned int r = x + 0x7FFFu + ((x >> 16) & 1u);   // round-to-nearest-even
  return (unsigned short)(r >> 16);
}
__device__ __forceinline__ float bf16_bits_to_f32(unsigned short u) {
  unsigned int x = ((unsigned int)u) << 16;
  return __builtin_bit_cast(float, x);
}

union FragU {
  uint4 q[2];
  v16bf v;
};

// A-matrix 16x32 bf16 fragment: lane m = lane&15, hi = lane>>4.
// Elements 0..15 <-> k = 16*hi + i (contiguous, matches load_BT_vec)
__device__ __forceinline__ v16bf load_A_vec(const unsigned short* s, int ld, int lane) {
  int m = lane & 15;
  int hi = lane >> 4;
  const unsigned short* p = s + (size_t)m * ld + hi * 16;
  FragU u;
  u.q[0] = *(const uint4*)(p);
  u.q[1] = *(const uint4*)(p + 8);
  return u.v;
}

// B-matrix 32x16 fragment from an N-major (transposed, N x K row-major) source:
// lane n = lane&15, kb = (lane>>4)*16; elements i <-> B[kb+i][n] = s[n*ld+kb+i],
// 16 contiguous ushorts per lane.
__device__ __forceinline__ v16bf load_BT_vec(const unsigned short* s, int ld, int lane) {
  int n = lane & 15;
  int kb = (lane >> 4) * 16;
  const unsigned short* p = s + (size_t)n * ld + kb;
  FragU u;
  u.q[0] = *(const uint4*)(p);
  u.q[1] = *(const uint4*)(p + 8);
  return u.v;
}

__device__ __forceinline__ v8f wmma_bf16(v16bf a, v16bf b, v8f c) {
  return __builtin_amdgcn_wmma_f32_16x16x32_bf16(false, a, false, b, (short)0, c,
                                                 false, false);
}

// gfx1250 async global->LDS copy, 16 bytes per lane. Tracked by ASYNCcnt.
__device__ __forceinline__ void async_copy_b128(const void* gptr, void* lptr) {
  unsigned lds = (unsigned)(uintptr_t)lptr;           // low 32 bits = LDS offset
  unsigned long long ga = (unsigned long long)(uintptr_t)gptr;
  asm volatile("global_load_async_to_lds_b128 %0, %1, off"
               :: "v"(lds), "v"(ga) : "memory");
}
// Async ops retire in order: waiting ASYNCcnt<=N leaves only the newest N in
// flight, so everything issued before them is complete.
__device__ __forceinline__ void async_wait0()    { asm volatile("s_wait_asynccnt 0x0" ::: "memory"); }
__device__ __forceinline__ void async_wait_le6() { asm volatile("s_wait_asynccnt 0x6" ::: "memory"); }
__device__ __forceinline__ void async_wait_le8() { asm volatile("s_wait_asynccnt 0x8" ::: "memory"); }

// ------------------------------ kernels ------------------------------------

__global__ void cvt_f32_bf16_kernel(const float* __restrict__ in,
                                    unsigned short* __restrict__ out, size_t n) {
  size_t i = (size_t)blockIdx.x * blockDim.x + threadIdx.x;
  if (i < n) out[i] = f32_to_bf16_bits(in[i]);
}

// f32 [R,C] row-major -> bf16 [C,R] row-major (transpose + downconvert).
// block (32,8), grid (C/32, R/32).
__global__ void cvt_transpose_kernel(const float* __restrict__ in,
                                     unsigned short* __restrict__ out,
                                     int R, int C) {
  __shared__ float tile[32][33];
  int c0 = blockIdx.x * 32, r0 = blockIdx.y * 32;
  int tx = threadIdx.x, ty = threadIdx.y;
#pragma unroll
  for (int i = 0; i < 4; ++i) {
    int r = r0 + ty + i * 8;
    tile[ty + i * 8][tx] = in[(size_t)r * C + c0 + tx];
  }
  __syncthreads();
#pragma unroll
  for (int i = 0; i < 4; ++i) {
    int c = c0 + ty + i * 8;
    out[(size_t)c * R + r0 + tx] = f32_to_bf16_bits(tile[tx][ty + i * 8]);
  }
}

// Tiled GEMM: C[M,N] = A[M,K] @ B[K,N] + bias, A bf16 row-major, B given
// TRANSPOSED as BT[N,K] bf16 row-major. WG = 128 threads (4 waves), 128x64
// output tile (each wave 32x64 -> 8 WMMA per K-step), K-step 32, double-
// buffered async staging. vt_mode: scatter-store bf16 output as
// [row/S, col, row%S] (used to produce V^T for attention).
__global__ void gemm_wmma_bf16_kernel(const unsigned short* __restrict__ A,
                                      const unsigned short* __restrict__ BT,
                                      const float* __restrict__ bias,
                                      unsigned short* __restrict__ out_bf,
                                      float* __restrict__ out_f,
                                      int M, int N, int K, int vt_mode) {
  __shared__ __align__(16) unsigned short a_lds[2][128 * 32];
  __shared__ __align__(16) unsigned short bT_lds[2][64 * 32];

  const int tid = threadIdx.x;
  const int wave = tid >> 5;
  const int lane = tid & 31;
  const int m0 = blockIdx.y * 128;
  const int n0 = blockIdx.x * 64;

  const int brow = tid >> 1;         // 0..63
  const int bcg = (tid & 1) * 16;    // 0 or 16

  // 6 async b128 copies per thread per tile: A row (tid) 32 ushorts,
  // BT half-row 16 ushorts.
  auto stage = [&](int k0, int bufi) {
    const unsigned short* asrc = A + (size_t)(m0 + tid) * K + k0;
    unsigned short* ad = &a_lds[bufi][tid * 32];
    async_copy_b128(asrc,      ad);
    async_copy_b128(asrc + 8,  ad + 8);
    async_copy_b128(asrc + 16, ad + 16);
    async_copy_b128(asrc + 24, ad + 24);
    const unsigned short* bsrc = BT + (size_t)(n0 + brow) * K + k0 + bcg;
    unsigned short* bd = &bT_lds[bufi][brow * 32 + bcg];
    async_copy_b128(bsrc,     bd);
    async_copy_b128(bsrc + 8, bd + 8);
  };

  v8f acc[2][4] = {};

  const int nt = K / 32;
  stage(0, 0);

  for (int t = 0; t < nt; ++t) {
    const int cur = t & 1;
    if (t + 1 < nt) {
      stage((t + 1) * 32, cur ^ 1);   // overlap DMA of t+1 with compute of t
      async_wait_le6();               // tile t complete; t+1 still in flight
    } else {
      async_wait0();
    }
    if (t + 2 < nt) {
      __builtin_prefetch(A + (size_t)(m0 + tid) * K + (t + 2) * 32, 0, 1);
      __builtin_prefetch(BT + (size_t)(n0 + brow) * K + (t + 2) * 32 + bcg, 0, 1);
    }
    __syncthreads();

    v16bf a0 = load_A_vec(&a_lds[cur][(wave * 32) * 32], 32, lane);
    v16bf a1 = load_A_vec(&a_lds[cur][(wave * 32 + 16) * 32], 32, lane);
    v16bf bf[4];
#pragma unroll
    for (int c = 0; c < 4; ++c)
      bf[c] = load_BT_vec(&bT_lds[cur][(c * 16) * 32], 32, lane);
#pragma unroll
    for (int c = 0; c < 4; ++c) {
      acc[0][c] = wmma_bf16(a0, bf[c], acc[0][c]);
      acc[1][c] = wmma_bf16(a1, bf[c], acc[1][c]);
    }
    __syncthreads();
  }

  // epilogue: C layout (M = r + 8*hi, N = lane&15)
  const int hi = lane >> 4;
  const int nn = lane & 15;
#pragma unroll
  for (int mt = 0; mt < 2; ++mt) {
#pragma unroll
    for (int c = 0; c < 4; ++c) {
      int col = n0 + c * 16 + nn;
      float bv = bias ? bias[col] : 0.0f;
#pragma unroll
      for (int r = 0; r < 8; ++r) {
        int orow = m0 + wave * 32 + mt * 16 + r + 8 * hi;
        float v = acc[mt][c][r] + bv;
        if (out_bf) {
          size_t idx;
          if (vt_mode) {
            // [b, col, s] with b = orow/S, s = orow%S
            idx = ((size_t)(orow >> 11) * N + col) * (size_t)S_ + (orow & (S_ - 1));
          } else {
            idx = (size_t)orow * N + col;
          }
          out_bf[idx] = f32_to_bf16_bits(v);
        } else {
          out_f[(size_t)orow * N + col] = v;
        }
      }
    }
  }
}

// RoPE in-place on bf16 tensor laid out [B, S, nheads, HD].
__global__ void rope_kernel(unsigned short* __restrict__ x,
                            const int* __restrict__ pos_ids, int nheads,
                            size_t total_pairs) {
  size_t i = (size_t)blockIdx.x * blockDim.x + threadIdx.x;
  if (i >= total_pairs) return;
  const int half = HD_ / 2;  // 64
  int d = (int)(i % half);
  size_t t = i / half;
  int h = (int)(t % nheads);
  size_t t2 = t / nheads;
  int s = (int)(t2 % S_);
  int b = (int)(t2 / S_);

  float pos = (float)pos_ids[s];
  float freq = __powf(1.0e6f, -((float)d) / (float)half);
  float ang = pos * freq;
  float sn, cs;
  __sincosf(ang, &sn, &cs);

  size_t base = (((size_t)b * S_ + s) * nheads + h) * HD_;
  float x1 = bf16_bits_to_f32(x[base + d]);
  float x2 = bf16_bits_to_f32(x[base + d + half]);
  x[base + d]        = f32_to_bf16_bits(x1 * cs - x2 * sn);
  x[base + d + half] = f32_to_bf16_bits(x1 * sn + x2 * cs);
}

// Flash attention. Grid: (S/64, NH, B), block: 128 (4 waves).
// Q:[B,S,NH,HD] bf16 (post-RoPE), K:[B,S,NKV,HD] bf16 (post-RoPE),
// VT:[B,NKV,HD,S] bf16, Ctx:[B,S,NH,HD] bf16. Double-buffered K/V staging.
__global__ void flash_attn_kernel(const unsigned short* __restrict__ Q,
                                  const unsigned short* __restrict__ Kc,
                                  const unsigned short* __restrict__ VT,
                                  unsigned short* __restrict__ Ctx) {
  __shared__ __align__(16) unsigned short k_lds[2][32 * HD_];   // [key][dim]
  __shared__ __align__(16) unsigned short vT_lds[2][HD_ * 32];  // [dim][key]
  __shared__ __align__(16) unsigned short p_lds[4][16 * 32];

  const int tid = threadIdx.x;
  const int wave = tid >> 5;
  const int lane = tid & 31;
  const int b = blockIdx.z;
  const int h = blockIdx.y;
  const int qb0 = blockIdx.x * 64;
  const int qbase = qb0 + wave * 16;
  const int kvh = h / (NH_ / NKV_);
  const float scale = 0.08838834764831845f;  // 1/sqrt(128)

  // Q tile 16x128 in A layout, 4 K-chunks of 32 (vectorized global loads)
  v16bf qa[4];
  {
    const unsigned short* qptr =
        Q + ((size_t)b * S_ + qbase) * (NH_ * HD_) + (size_t)h * HD_;
#pragma unroll
    for (int kk = 0; kk < 4; ++kk)
      qa[kk] = load_A_vec(qptr + kk * 32, NH_ * HD_, lane);
  }

  v8f o[8] = {};
  float mrow[8], lrow[8];
#pragma unroll
  for (int r = 0; r < 8; ++r) { mrow[r] = -1.0e30f; lrow[r] = 0.0f; }

  const int hi = lane >> 4;
  const int nn = lane & 15;
  const int nj = qb0 / 32 + 2;  // key blocks of 32, causal bound for this WG

  const int krow = tid >> 2;              // key row 0..31
  const int kcg = (tid & 3) * 32;         // dim chunk
  const unsigned short* vt_base =
      VT + ((size_t)(b * NKV_ + kvh) * HD_ + tid) * S_;  // dim = tid (0..127)

  // 8 async b128 copies per thread per key-block.
  auto stage = [&](int jj, int bufi) {
    const int kb = jj * 32;
    const unsigned short* ks =
        Kc + ((size_t)b * S_ + kb + krow) * (NKV_ * HD_) + (size_t)kvh * HD_ + kcg;
    unsigned short* kd = &k_lds[bufi][krow * HD_ + kcg];
    async_copy_b128(ks,      kd);
    async_copy_b128(ks + 8,  kd + 8);
    async_copy_b128(ks + 16, kd + 16);
    async_copy_b128(ks + 24, kd + 24);
    const unsigned short* vs = vt_base + kb;
    unsigned short* vd = &vT_lds[bufi][tid * 32];
    async_copy_b128(vs,      vd);
    async_copy_b128(vs + 8,  vd + 8);
    async_copy_b128(vs + 16, vd + 16);
    async_copy_b128(vs + 24, vd + 24);
  };

  stage(0, 0);

  for (int j = 0; j < nj; ++j) {
    const int cur = j & 1;
    const int kb0 = j * 32;
    if (j + 1 < nj) {
      stage(j + 1, cur ^ 1);   // overlap DMA of block j+1 with compute of j
      async_wait_le8();
    } else {
      async_wait0();
    }
    __syncthreads();

    // S tile 16x32 = two 16x16 halves; B = K^T chunks from k_lds (key-major)
    v8f sacc[2] = {};
#pragma unroll
    for (int kk = 0; kk < 4; ++kk) {
#pragma unroll
      for (int t2 = 0; t2 < 2; ++t2) {
        v16bf bkt =
            load_BT_vec(&k_lds[cur][(size_t)t2 * 16 * HD_ + kk * 32], HD_, lane);
        sacc[t2] = wmma_bf16(qa[kk], bkt, sacc[t2]);
      }
    }

    // scale + causal mask + online softmax
#pragma unroll
    for (int r = 0; r < 8; ++r) {
      int qrow = qbase + r + 8 * hi;
#pragma unroll
      for (int t2 = 0; t2 < 2; ++t2) {
        int col = kb0 + t2 * 16 + nn;
        float v = sacc[t2][r] * scale;
        sacc[t2][r] = (col > qrow) ? -1.0e30f : v;
      }
      float mx = fmaxf(sacc[0][r], sacc[1][r]);
#pragma unroll
      for (int dmask = 1; dmask < 16; dmask <<= 1)
        mx = fmaxf(mx, __shfl_xor(mx, dmask, 32));
      float mnew = fmaxf(mrow[r], mx);
      float corr = __expf(mrow[r] - mnew);
      mrow[r] = mnew;
      lrow[r] *= corr;
#pragma unroll
      for (int c = 0; c < 8; ++c) o[c][r] *= corr;
      float psum = 0.0f;
#pragma unroll
      for (int t2 = 0; t2 < 2; ++t2) {
        float p = __expf(sacc[t2][r] - mnew);
        sacc[t2][r] = p;
        psum += p;
      }
#pragma unroll
      for (int dmask = 1; dmask < 16; dmask <<= 1)
        psum += __shfl_xor(psum, dmask, 32);
      lrow[r] += psum;
    }

    // round-trip P through LDS to re-stripe into A layout (bf16 16x32)
#pragma unroll
    for (int r = 0; r < 8; ++r) {
#pragma unroll
      for (int t2 = 0; t2 < 2; ++t2) {
        p_lds[wave][(r + 8 * hi) * 32 + t2 * 16 + nn] = f32_to_bf16_bits(sacc[t2][r]);
      }
    }
    // same-wave LDS ops are in-order; p_lds is wave-private
    v16bf pa = load_A_vec(p_lds[wave], 32, lane);
#pragma unroll
    for (int c = 0; c < 8; ++c) {
      v16bf vb = load_BT_vec(&vT_lds[cur][(size_t)(c * 16) * 32], 32, lane);
      o[c] = wmma_bf16(pa, vb, o[c]);
    }
    __syncthreads();
  }

  // finalize: divide by l, store context bf16 [B,S,NH,HD] -> flat [B,S,H]
#pragma unroll
  for (int r = 0; r < 8; ++r) {
    float inv = 1.0f / lrow[r];
    int qrow = qbase + r + 8 * hi;
#pragma unroll
    for (int c = 0; c < 8; ++c) {
      float v = o[c][r] * inv;
      Ctx[((size_t)b * S_ + qrow) * H_ + (size_t)h * HD_ + c * 16 + nn] =
          f32_to_bf16_bits(v);
    }
  }
}

// ------------------------------ launch -------------------------------------

extern "C" void kernel_launch(void* const* d_in, const int* in_sizes, int n_in,
                              void* d_out, int out_size, void* d_ws, size_t ws_size,
                              hipStream_t stream) {
  (void)in_sizes; (void)n_in; (void)out_size; (void)ws_size;

  const float* hidden = (const float*)d_in[0];
  // d_in[1] = attention_mask (causal; computed analytically, not read)
  const int* pos_ids = (const int*)d_in[2];
  const float* Wq = (const float*)d_in[3];
  const float* bq = (const float*)d_in[4];
  const float* Wk = (const float*)d_in[5];
  const float* bk = (const float*)d_in[6];
  const float* Wv = (const float*)d_in[7];
  const float* bv = (const float*)d_in[8];
  const float* Wo = (const float*)d_in[9];
  float* out = (float*)d_out;

  unsigned short* wsb = (unsigned short*)d_ws;
  size_t off = 0;
  unsigned short* Xbf  = wsb + off; off += (size_t)MROWS_ * H_;   // X bf16
  unsigned short* WqT  = wsb + off; off += (size_t)H_ * H_;       // Wq^T bf16
  unsigned short* WkT  = wsb + off; off += (size_t)H_ * KV_;      // Wk^T bf16
  unsigned short* WvT  = wsb + off; off += (size_t)H_ * KV_;      // Wv^T bf16
  unsigned short* WoT  = wsb + off; off += (size_t)H_ * H_;       // Wo^T bf16
  unsigned short* Qb   = wsb + off; off += (size_t)MROWS_ * H_;   // Q bf16
  unsigned short* Kb   = wsb + off; off += (size_t)MROWS_ * KV_;  // K bf16
  unsigned short* VTb  = wsb + off; off += (size_t)MROWS_ * KV_;  // V^T bf16
  unsigned short* Ctx  = wsb + off; off += (size_t)MROWS_ * H_;   // context bf16

  // 1) downconvert X; downconvert + transpose weights
  {
    size_t n = (size_t)MROWS_ * H_;
    cvt_f32_bf16_kernel<<<(int)((n + 255) / 256), 256, 0, stream>>>(hidden, Xbf, n);

    dim3 tb(32, 8);
    cvt_transpose_kernel<<<dim3(H_ / 32, H_ / 32), tb, 0, stream>>>(Wq, WqT, H_, H_);
    cvt_transpose_kernel<<<dim3(KV_ / 32, H_ / 32), tb, 0, stream>>>(Wk, WkT, H_, KV_);
    cvt_transpose_kernel<<<dim3(KV_ / 32, H_ / 32), tb, 0, stream>>>(Wv, WvT, H_, KV_);
    cvt_transpose_kernel<<<dim3(H_ / 32, H_ / 32), tb, 0, stream>>>(Wo, WoT, H_, H_);
  }

  // 2) QKV projections (bf16 out, f32 accum, +bias); V written transposed
  {
    dim3 blk(128);
    gemm_wmma_bf16_kernel<<<dim3(H_ / 64, MROWS_ / 128), blk, 0, stream>>>(
        Xbf, WqT, bq, Qb, nullptr, MROWS_, H_, H_, 0);
    gemm_wmma_bf16_kernel<<<dim3(KV_ / 64, MROWS_ / 128), blk, 0, stream>>>(
        Xbf, WkT, bk, Kb, nullptr, MROWS_, KV_, H_, 0);
    gemm_wmma_bf16_kernel<<<dim3(KV_ / 64, MROWS_ / 128), blk, 0, stream>>>(
        Xbf, WvT, bv, VTb, nullptr, MROWS_, KV_, H_, 1);
  }

  // 3) RoPE on Q and K
  {
    size_t pq = (size_t)B_ * S_ * NH_ * (HD_ / 2);
    size_t pk = (size_t)B_ * S_ * NKV_ * (HD_ / 2);
    rope_kernel<<<(int)((pq + 255) / 256), 256, 0, stream>>>(Qb, pos_ids, NH_, pq);
    rope_kernel<<<(int)((pk + 255) / 256), 256, 0, stream>>>(Kb, pos_ids, NKV_, pk);
  }

  // 4) fused flash attention
  {
    flash_attn_kernel<<<dim3(S_ / 64, NH_, B_), dim3(128), 0, stream>>>(Qb, Kb, VTb,
                                                                        Ctx);
  }

  // 5) output projection (f32 out to d_out)
  {
    gemm_wmma_bf16_kernel<<<dim3(H_ / 64, MROWS_ / 128), dim3(128), 0, stream>>>(
        Ctx, WoT, nullptr, nullptr, out, MROWS_, H_, H_, 0);
  }
}
